// QKVMultiheadCrossAttention_86603720556579
// MI455X (gfx1250) — matrix operand order repaired and checked by reading, not compiled, not run on hardware
//
#include <hip/hip_runtime.h>
#include <hip/hip_bf16.h>

#define HEADS 16
#define CH 64
#define NCTX 2048
#define NDATA 2048
#define BSZ 2
#define BM 128        // query rows per block (8 waves x 16)
#define BN 64         // kv rows per tile
#define STRIDE 72     // f16 row stride (144B, 16B aligned)

typedef _Float16 v16h __attribute__((ext_vector_type(16)));
typedef _Float16 v8h  __attribute__((ext_vector_type(8)));
typedef float    v8f  __attribute__((ext_vector_type(8)));

// k side: 1/sqrt(sqrt(64));  q side additionally folds log2(e) so softmax uses raw exp2
#define KSCALE 0.3535533905932738f
#define QSCALE 0.51013381089582764f   // KSCALE * log2(e)

#if defined(__has_builtin) && __has_builtin(__builtin_amdgcn_s_wait_dscnt)
#define WAIT_DS() __builtin_amdgcn_s_wait_dscnt(0)
#else
#define WAIT_DS() asm volatile("s_wait_dscnt 0" ::: "memory")
#endif

#if defined(__has_builtin) && __has_builtin(__builtin_amdgcn_exp2f)
#define EXP2F(x) __builtin_amdgcn_exp2f(x)
#else
#define EXP2F(x) exp2f(x)
#endif

// ---- single-instruction DPP reductions: op applies DPP to src0, plain src1 ----
// rotate-reduce over a 16-lane DPP row (lanes 0-15 / 16-31); all lanes get result
__device__ __forceinline__ float rowmax16(float x) {
  asm("v_max_num_f32_dpp %0, %0, %0 row_ror:1 row_mask:0xf bank_mask:0xf bound_ctrl:1" : "+v"(x));
  asm("v_max_num_f32_dpp %0, %0, %0 row_ror:2 row_mask:0xf bank_mask:0xf bound_ctrl:1" : "+v"(x));
  asm("v_max_num_f32_dpp %0, %0, %0 row_ror:4 row_mask:0xf bank_mask:0xf bound_ctrl:1" : "+v"(x));
  asm("v_max_num_f32_dpp %0, %0, %0 row_ror:8 row_mask:0xf bank_mask:0xf bound_ctrl:1" : "+v"(x));
  return x;
}
__device__ __forceinline__ float rowsum16(float x) {
  asm("v_add_f32_dpp %0, %0, %0 row_ror:1 row_mask:0xf bank_mask:0xf bound_ctrl:1" : "+v"(x));
  asm("v_add_f32_dpp %0, %0, %0 row_ror:2 row_mask:0xf bank_mask:0xf bound_ctrl:1" : "+v"(x));
  asm("v_add_f32_dpp %0, %0, %0 row_ror:4 row_mask:0xf bank_mask:0xf bound_ctrl:1" : "+v"(x));
  asm("v_add_f32_dpp %0, %0, %0 row_ror:8 row_mask:0xf bank_mask:0xf bound_ctrl:1" : "+v"(x));
  return x;
}
// sum over 4-lane quads via quad_perm rotations
__device__ __forceinline__ void quadsum2(float& a, float& b) {
  asm("v_add_f32_dpp %0, %0, %0 quad_perm:[1,2,3,0] row_mask:0xf bank_mask:0xf bound_ctrl:1" : "+v"(a));
  asm("v_add_f32_dpp %0, %0, %0 quad_perm:[1,2,3,0] row_mask:0xf bank_mask:0xf bound_ctrl:1" : "+v"(b));
  asm("v_add_f32_dpp %0, %0, %0 quad_perm:[2,3,0,1] row_mask:0xf bank_mask:0xf bound_ctrl:1" : "+v"(a));
  asm("v_add_f32_dpp %0, %0, %0 quad_perm:[2,3,0,1] row_mask:0xf bank_mask:0xf bound_ctrl:1" : "+v"(b));
}

__device__ __forceinline__ v8f wmma16(v16h a, v16h b, v8f c) {
  return __builtin_amdgcn_wmma_f32_16x16x32_f16(false, a, false, b, (short)0, c,
                                                false, false);
}
// A-fragment: elems 0..7 = p[0..7], elems 8..15 = p[16..23]
__device__ __forceinline__ v16h ld_frag_split(const _Float16* p) {
  v16h r;
  ((v8h*)&r)[0] = *(const v8h*)p;
  ((v8h*)&r)[1] = *(const v8h*)(p + 16);
  return r;
}
// B-fragment: elems 0..7 = p[0..7], elems 8..15 = p[16..23]
__device__ __forceinline__ v16h ld_frag_contig(const _Float16* p) {
  v16h r;
  ((v8h*)&r)[0] = *(const v8h*)p;
  ((v8h*)&r)[1] = *(const v8h*)(p + 16);
  return r;
}

__global__ __launch_bounds__(256) void flash_xattn(
    const float* __restrict__ q, const float* __restrict__ kv,
    const float* __restrict__ qw, const float* __restrict__ qb,
    const float* __restrict__ kw, const float* __restrict__ kb,
    float* __restrict__ out)
{
  __shared__ __align__(16) _Float16 smem[18432];   // 36.9 KB
  _Float16* q_sm  = smem;                  // [128][72] prologue only (overlaid)
  _Float16* k_sm  = smem;                  // [64][72]
  _Float16* vt_sm = smem + 64 * STRIDE;    // [64][72]  v transposed: [c][s]
  _Float16* p_sm  = smem + 128 * STRIDE;   // [8 waves][16][72]

  const int tid  = threadIdx.x;
  const int lane = tid & 31;
  const int wave = tid >> 5;
  const int m    = lane & 15;
  const int half = lane >> 4;

  const int tt = blockIdx.x & 15;
  const int h  = (blockIdx.x >> 4) & 15;
  const int b  = blockIdx.x >> 8;
  const int t0 = tt * BM;

  // ---------- stage Q: layernorm + scale(+log2e) + cvt f16 ----------
  {
    const int sub = tid & 3;               // 4 lanes/row, 16 elems each
    const int rg  = tid >> 2;              // 0..63
    #pragma unroll
    for (int rr = 0; rr < BM; rr += 64) {
      const int row = rr + rg;
      const float* src = q + (((size_t)b * NCTX + t0 + row) * HEADS + h) * CH + sub * 16;
      float x[16]; float s1 = 0.f, s2 = 0.f;
      #pragma unroll
      for (int i = 0; i < 16; i++) { x[i] = src[i]; s1 += x[i]; s2 += x[i] * x[i]; }
      quadsum2(s1, s2);
      const float mu = s1 * (1.f / CH);
      const float rs = rsqrtf(s2 * (1.f / CH) - mu * mu + 1e-6f);
      #pragma unroll
      for (int i = 0; i < 16; i++) {
        const int c = sub * 16 + i;
        q_sm[row * STRIDE + c] = (_Float16)(((x[i] - mu) * rs * qw[c] + qb[c]) * QSCALE);
      }
    }
  }
  __syncthreads();

  // per-wave Q A-fragments, kept in registers for the whole kernel
  const _Float16* qbase = q_sm + (wave * 16 + m) * STRIDE + half * 8;
  v16h aq0 = ld_frag_split(qbase);        // K 0..31
  v16h aq1 = ld_frag_split(qbase + 32);   // K 32..63
  WAIT_DS();   // fragments captured before loop buffers overwrite q_sm

  float mrow[8], lpart[8];
  v8f o0 = {}, o1 = {}, o2 = {}, o3 = {};
  #pragma unroll
  for (int r = 0; r < 8; r++) { mrow[r] = -1e30f; lpart[r] = 0.f; }

  _Float16* pw = p_sm + wave * 16 * STRIDE;

  for (int s0 = 0; s0 < NDATA; s0 += BN) {
    __syncthreads();
    // ---------- stage K (layernorm+scale) and V (transposed f16) ----------
    {
      const int sub = tid & 3;
      const int row = tid >> 2;            // 0..63
      const float* ksrc = kv + (((size_t)b * NDATA + s0 + row) * HEADS + h) * (2 * CH) + sub * 16;
      float x[16]; float s1 = 0.f, s2 = 0.f;
      #pragma unroll
      for (int i = 0; i < 16; i++) { x[i] = ksrc[i]; s1 += x[i]; s2 += x[i] * x[i]; }
      quadsum2(s1, s2);
      const float mu = s1 * (1.f / CH);
      const float rs = rsqrtf(s2 * (1.f / CH) - mu * mu + 1e-6f);
      #pragma unroll
      for (int i = 0; i < 16; i++) {
        const int c = sub * 16 + i;
        k_sm[row * STRIDE + c] = (_Float16)(((x[i] - mu) * rs * kw[c] + kb[c]) * KSCALE);
      }
      const float* vsrc = ksrc + CH;
      #pragma unroll
      for (int i = 0; i < 16; i++)
        vt_sm[(sub * 16 + i) * STRIDE + row] = (_Float16)vsrc[i];
      if (s0 + BN < NDATA)
        __builtin_prefetch(ksrc + (size_t)BN * HEADS * 2 * CH, 0, 1);  // global_prefetch_b8
    }
    __syncthreads();

    // ---------- QK^T: 16x64 logits (4 tiles x K=64), base-2 domain ----------
    v8f w0 = {}, w1 = {}, w2 = {}, w3 = {};
    {
      const _Float16* kb0 = k_sm + m * STRIDE + half * 8;
      const _Float16* kb1 = kb0 + 16 * STRIDE;
      const _Float16* kb2 = kb0 + 32 * STRIDE;
      const _Float16* kb3 = kb0 + 48 * STRIDE;
      w0 = wmma16(aq0, ld_frag_contig(kb0),      w0);
      w0 = wmma16(aq1, ld_frag_contig(kb0 + 32), w0);
      w1 = wmma16(aq0, ld_frag_contig(kb1),      w1);
      w1 = wmma16(aq1, ld_frag_contig(kb1 + 32), w1);
      w2 = wmma16(aq0, ld_frag_contig(kb2),      w2);
      w2 = wmma16(aq1, ld_frag_contig(kb2 + 32), w2);
      w3 = wmma16(aq0, ld_frag_contig(kb3),      w3);
      w3 = wmma16(aq1, ld_frag_contig(kb3 + 32), w3);
    }

    // ---------- online softmax: DPP row-max, exp2, per-lane partial sums ----------
    #pragma unroll
    for (int r = 0; r < 8; r++) {
      float tm = fmaxf(fmaxf(w0[r], w1[r]), fmaxf(w2[r], w3[r]));
      tm = rowmax16(tm);
      const float mnew  = fmaxf(mrow[r], tm);
      const float alpha = EXP2F(mrow[r] - mnew);
      const float p0 = EXP2F(w0[r] - mnew);
      const float p1 = EXP2F(w1[r] - mnew);
      const float p2 = EXP2F(w2[r] - mnew);
      const float p3 = EXP2F(w3[r] - mnew);
      lpart[r] = lpart[r] * alpha + ((p0 + p1) + (p2 + p3));  // reduced in epilogue
      mrow[r]  = mnew;
      o0[r] *= alpha; o1[r] *= alpha; o2[r] *= alpha; o3[r] *= alpha;
      _Float16* pr = pw + (r + half * 8) * STRIDE + m;
      pr[0]  = (_Float16)p0;
      pr[16] = (_Float16)p1;
      pr[32] = (_Float16)p2;
      pr[48] = (_Float16)p3;
    }
    WAIT_DS();

    // ---------- P x V: K=64 via 2 WMMAs per output c-tile ----------
    v16h pa0 = ld_frag_split(pw + m * STRIDE + half * 8);
    v16h pa1 = ld_frag_split(pw + m * STRIDE + half * 8 + 32);
    const _Float16* vb = vt_sm + m * STRIDE + half * 8;
    o0 = wmma16(pa0, ld_frag_contig(vb),                    o0);
    o0 = wmma16(pa1, ld_frag_contig(vb + 32),               o0);
    o1 = wmma16(pa0, ld_frag_contig(vb + 16 * STRIDE),      o1);
    o1 = wmma16(pa1, ld_frag_contig(vb + 16 * STRIDE + 32), o1);
    o2 = wmma16(pa0, ld_frag_contig(vb + 32 * STRIDE),      o2);
    o2 = wmma16(pa1, ld_frag_contig(vb + 32 * STRIDE + 32), o2);
    o3 = wmma16(pa0, ld_frag_contig(vb + 48 * STRIDE),      o3);
    o3 = wmma16(pa1, ld_frag_contig(vb + 48 * STRIDE + 32), o3);
  }

  // ---------- epilogue: reduce softmax denominator once, normalize, store ----------
  #pragma unroll
  for (int r = 0; r < 8; r++) {
    const float inv = 1.f / rowsum16(lpart[r]);
    const size_t trow = (size_t)b * NCTX + t0 + wave * 16 + half * 8 + r;
    float* dst = out + (trow * HEADS + h) * CH + m;
    dst[0]  = o0[r] * inv;
    dst[16] = o1[r] * inv;
    dst[32] = o2[r] * inv;
    dst[48] = o3[r] * inv;
  }
}

extern "C" void kernel_launch(void* const* d_in, const int* in_sizes, int n_in,
                              void* d_out, int out_size, void* d_ws, size_t ws_size,
                              hipStream_t stream) {
  const float* q  = (const float*)d_in[0];
  const float* kv = (const float*)d_in[1];
  const float* qw = (const float*)d_in[2];
  const float* qb = (const float*)d_in[3];
  const float* kw = (const float*)d_in[4];
  const float* kb = (const float*)d_in[5];
  float* out = (float*)d_out;

  dim3 grid(BSZ * HEADS * (NCTX / BM));  // 512 blocks
  flash_xattn<<<grid, 256, 0, stream>>>(q, kv, qw, qb, kw, kb, out);
}
